// Model_56504589746531
// MI455X (gfx1250) — compile-verified, matrix operand order checked by reading, not run on hardware
//
#include <hip/hip_runtime.h>
#include <hip/hip_bf16.h>
#include <math.h>

typedef float  v8f  __attribute__((ext_vector_type(8)));
typedef __bf16 v16bf __attribute__((ext_vector_type(16)));
typedef unsigned int v8u __attribute__((ext_vector_type(8)));

#define N_NODES 10000
#define N_EDGES 160000
#define CUTR 2.0f
#define INV_AVG (1.0f/3.0f)
#define INV_SQRT2 0.70710678118654752f
#define PI_F 3.14159265358979f

// ---- workspace layout (float offsets) ----
#define WS_U    0u            // [E*3]
#define WS_ENV  480000u       // [E]
#define WS_X    640000u       // [E*64]
#define WS_V    10880000u     // [E*3*64]
#define WS_RHO0 41600000u     // [N*64]
#define WS_RHO1 42240000u     // [N*3*64]
#define WS_WB   44160000u     // packed bf16 weights (as u32 pairs) start here
// u32 offsets inside weight area; one B tile = 32x16 bf16 = 256 u32
#define WP_EMB1 0u
#define WP_EMB2 8192u
#define WP_WV   24576u
#define WP_WENV(l) (26624u + (unsigned)(l)*2048u)
#define WP_LAT1(l) (30720u + (unsigned)(l)*49152u)
#define WP_LAT2(l) (129024u + (unsigned)(l)*16384u)
#define WP_MIX(l)  (161792u + (unsigned)(l)*4096u)

// ---------------- helpers ----------------
__device__ __forceinline__ unsigned short bfb(float f) {
  union { float f; unsigned u; } a; a.f = f;
  unsigned r = a.u + 0x7FFFu + ((a.u >> 16) & 1u);   // round-to-nearest-even
  return (unsigned short)(r >> 16);
}
__device__ __forceinline__ unsigned pk2(float lo, float hi) {
  return (unsigned)bfb(lo) | ((unsigned)bfb(hi) << 16);
}
// A-layout (16x32 bf16, wave32): lane<16 holds K {0..7,16..23}, lane>=16 holds {8..15,24..31}
// a_pair_k returns the even K of VGPR j's pair.
__device__ __forceinline__ int a_pair_k(int lane, int j) {
  int b = (j < 4) ? (2 * j) : (16 + 2 * (j - 4));
  return b + ((lane & 16) ? 8 : 0);
}
// C/D f32 16x16: VGPR r holds row r (lanes 0-15) / row r+8 (lanes 16-31), col = lane&15
__device__ __forceinline__ int cm_row(int r, int lane) { return r + ((lane & 16) ? 8 : 0); }

// B tile: 256 u32 per tile, lane-contiguous 8 u32 (two b128 loads)
__device__ __forceinline__ v16bf load_B(const unsigned int* wp, int tile, int lane) {
  const v8u* p = (const v8u*)(wp + (size_t)tile * 256 + lane * 8);
  return __builtin_bit_cast(v16bf, *p);
}
// A from u32-typed LDS holding packed bf16 pairs; stride32 = row stride in u32
__device__ __forceinline__ v16bf load_A_ldsu(const unsigned int* lds, int stride32, int k0, int lane) {
  int m = lane & 15;
  v8u v;
#pragma unroll
  for (int j = 0; j < 8; ++j) {
    int k = k0 + a_pair_k(lane, j);          // always even
    v[j] = lds[m * stride32 + (k >> 1)];
  }
  return __builtin_bit_cast(v16bf, v);
}
// A from float-typed LDS (convert to bf16 at load); stride = row stride in floats
__device__ __forceinline__ v16bf load_A_ldsf(const float* lds, int stride, int k0, int lane) {
  int m = lane & 15;
  v8u v;
#pragma unroll
  for (int j = 0; j < 8; ++j) {
    int k = k0 + a_pair_k(lane, j);
    v[j] = pk2(lds[m * stride + k], lds[m * stride + k + 1]);
  }
  return __builtin_bit_cast(v16bf, v);
}
__device__ __forceinline__ v8f wmma_bf(v16bf a, v16bf b, v8f c) {
  return __builtin_amdgcn_wmma_f32_16x16x32_bf16(false, a, false, b, (short)0, c, false, false);
}
// fast silu: v * rcp(1+exp(-v)) -- single v_rcp_f32 instead of IEEE div sequence
__device__ __forceinline__ float silu(float v) {
  return v * __builtin_amdgcn_rcpf(1.0f + __expf(-v));
}

// -------- weight pre-pack: row-major f32 [K,N] -> bf16 B-tiles (32x16) as u32 pairs --------
__global__ void k_pack(const float* __restrict__ W, int K, int N,
                       unsigned int* __restrict__ dst, int ntiles) {
  int tid = blockIdx.x * blockDim.x + threadIdx.x;
  if (tid >= ntiles * 32) return;
  int lane = tid & 31, tile = tid >> 5;
  int ntc = N >> 4;
  int kt = tile / ntc, nt = tile % ntc;
  int n = nt * 16 + (lane & 15);
#pragma unroll
  for (int j = 0; j < 8; ++j) {
    int k = kt * 32 + a_pair_k(lane, j);
    float v0 = (k < K) ? W[(size_t)k * N + n] : 0.0f;
    float v1 = (k + 1 < K) ? W[(size_t)(k + 1) * N + n] : 0.0f;
    dst[(size_t)tile * 256 + lane * 8 + j] = pk2(v0, v1);
  }
}

__global__ void k_zero(float* p, int n) {
  int i = blockIdx.x * blockDim.x + threadIdx.x;
  if (i < n) p[i] = 0.0f;
}

__device__ __forceinline__ float featval(int k, float xc, float bscale, int ssp, int rsp) {
  if (k < 8)   return __sinf((float)(k + 1) * PI_F * xc) * bscale; // bessel * env
  if (k == 8)  return (ssp == 0) ? 1.0f : 0.0f;
  if (k == 9)  return (ssp == 1) ? 1.0f : 0.0f;
  if (k == 10) return (rsp == 0) ? 1.0f : 0.0f;
  if (k == 11) return (rsp == 1) ? 1.0f : 0.0f;
  return 0.0f;
}

// -------- embed: geometry + two-body MLP + initial V --------
#define HSTF 36   // float stride, 16 rows
#define XSTF 68
__global__ void __launch_bounds__(128) k_embed(
    const float* __restrict__ pos, const int* __restrict__ species,
    const int* __restrict__ senders, const int* __restrict__ receivers,
    const unsigned int* __restrict__ wb, float* __restrict__ ws) {
  __shared__ float hstage[4][16 * HSTF];
  __shared__ float xstage[4][16 * XSTF];
  __shared__ float env_s[4][16];
  __shared__ float u_s[4][16][3];
  int lane = threadIdx.x & 31, wv = threadIdx.x >> 5;
  int e0 = (blockIdx.x * 4 + wv) * 16;
  int m = lane & 15, e = e0 + m;

  int si = senders[e], ri = receivers[e];
  float rx = pos[ri * 3 + 0] - pos[si * 3 + 0];
  float ry = pos[ri * 3 + 1] - pos[si * 3 + 1];
  float rz = pos[ri * 3 + 2] - pos[si * 3 + 2];
  float d = sqrtf(rx * rx + ry * ry + rz * rz);
  float dsafe = fmaxf(d, 1e-6f);
  float inv_d = __builtin_amdgcn_rcpf(dsafe);
  float ux = rx * inv_d, uy = ry * inv_d, uz = rz * inv_d;
  float xc = d * (1.0f / CUTR);
  float t1 = 1.0f - xc * xc;
  float env = (xc < 1.0f) ? t1 * t1 : 0.0f;
  int ssp = species[si], rsp = species[ri];
  if (lane < 16) {
    ws[WS_U + (size_t)e * 3 + 0] = ux;
    ws[WS_U + (size_t)e * 3 + 1] = uy;
    ws[WS_U + (size_t)e * 3 + 2] = uz;
    ws[WS_ENV + e] = env;
    env_s[wv][m] = env;
    u_s[wv][m][0] = ux; u_s[wv][m][1] = uy; u_s[wv][m][2] = uz;
  }
  // feat tile in A layout (K padded 12 -> 32 with zeros)
  float bscale = sqrtf(2.0f / CUTR) * inv_d * env;
  v8u fv;
#pragma unroll
  for (int j = 0; j < 8; ++j) {
    int k = a_pair_k(lane, j);
    fv[j] = pk2(featval(k, xc, bscale, ssp, rsp), featval(k + 1, xc, bscale, ssp, rsp));
  }
  v16bf a_feat = __builtin_bit_cast(v16bf, fv);

  v8f xacc[4] = {};
  for (int t = 0; t < 16; ++t) {
#pragma unroll
    for (int p = 0; p < 2; ++p) {
      v8f h = {};
      h = wmma_bf(a_feat, load_B(wb + WP_EMB1, 2 * t + p, lane), h);
#pragma unroll
      for (int r = 0; r < 8; ++r)
        hstage[wv][cm_row(r, lane) * HSTF + p * 16 + m] = silu(h[r]);
    }
    v16bf a_h = load_A_ldsf(hstage[wv], HSTF, 0, lane);
#pragma unroll
    for (int o = 0; o < 4; ++o)
      xacc[o] = wmma_bf(a_h, load_B(wb + WP_EMB2, t * 4 + o, lane), xacc[o]);
  }
  float evs[8];
#pragma unroll
  for (int r = 0; r < 8; ++r) evs[r] = env_s[wv][cm_row(r, lane)];
#pragma unroll
  for (int o = 0; o < 4; ++o)
#pragma unroll
    for (int r = 0; r < 8; ++r) {
      int mm = cm_row(r, lane);
      float xv = xacc[o][r] * evs[r];
      ws[WS_X + (size_t)(e0 + mm) * 64 + o * 16 + m] = xv;
      xstage[wv][mm * XSTF + o * 16 + m] = xv;
    }
  // V = (x @ W_v) outer u
  v8f vacc[4] = {};
#pragma unroll
  for (int kt = 0; kt < 2; ++kt) {
    v16bf a_x = load_A_ldsf(xstage[wv], XSTF, kt * 32, lane);
#pragma unroll
    for (int o = 0; o < 4; ++o)
      vacc[o] = wmma_bf(a_x, load_B(wb + WP_WV, kt * 4 + o, lane), vacc[o]);
  }
#pragma unroll
  for (int r = 0; r < 8; ++r) {
    int mm = cm_row(r, lane);
    float u3[3] = { u_s[wv][mm][0], u_s[wv][mm][1], u_s[wv][mm][2] };
#pragma unroll
    for (int o = 0; o < 4; ++o)
#pragma unroll
      for (int i = 0; i < 3; ++i)
        ws[WS_V + ((size_t)(e0 + mm) * 3 + i) * 64 + o * 16 + m] = vacc[o][r] * u3[i];
  }
}

// -------- env density: w_e = x @ W_env, atomic scatter to rho0/rho1 --------
__global__ void __launch_bounds__(128) k_env(
    const float* __restrict__ xg, const float* __restrict__ ug,
    const int* __restrict__ senders, const unsigned int* __restrict__ wenv,
    float* __restrict__ rho0, float* __restrict__ rho1) {
  __shared__ int s_s[4][16];
  __shared__ float u_s[4][16][3];
  int lane = threadIdx.x & 31, wv = threadIdx.x >> 5;
  int e0 = (blockIdx.x * 4 + wv) * 16;
  int m = lane & 15, e = e0 + m;
  if (lane < 16) {
    s_s[wv][m] = senders[e];
    u_s[wv][m][0] = ug[(size_t)e * 3 + 0];
    u_s[wv][m][1] = ug[(size_t)e * 3 + 1];
    u_s[wv][m][2] = ug[(size_t)e * 3 + 2];
  }
  v8f acc[4] = {};
#pragma unroll
  for (int kt = 0; kt < 2; ++kt) {
    v8u v;
#pragma unroll
    for (int j = 0; j < 8; ++j) {
      int k = kt * 32 + a_pair_k(lane, j);
      const float* xp = xg + (size_t)e * 64 + k;
      v[j] = pk2(xp[0], xp[1]);
    }
    v16bf a = __builtin_bit_cast(v16bf, v);
#pragma unroll
    for (int o = 0; o < 4; ++o)
      acc[o] = wmma_bf(a, load_B(wenv, kt * 4 + o, lane), acc[o]);
  }
#pragma unroll
  for (int r = 0; r < 8; ++r) {
    int mm = cm_row(r, lane);
    int s = s_s[wv][mm];
    float u3[3] = { u_s[wv][mm][0], u_s[wv][mm][1], u_s[wv][mm][2] };
#pragma unroll
    for (int o = 0; o < 4; ++o) {
      int c = o * 16 + m;
      float w = acc[o][r];
      atomicAdd(&rho0[(size_t)s * 64 + c], w);
#pragma unroll
      for (int i = 0; i < 3; ++i)
        atomicAdd(&rho1[((size_t)s * 3 + i) * 64 + c], w * u3[i]);
    }
  }
}

// -------- full layer: tensor products + fused lat MLP + mix + V update --------
#define LSTR32 100   // u32 row stride of lat (200 bf16)
__global__ void __launch_bounds__(128) k_layer(
    float* __restrict__ ws, const int* __restrict__ senders,
    const unsigned int* __restrict__ wlat1, const unsigned int* __restrict__ wlat2,
    const unsigned int* __restrict__ wmix) {
  __shared__ unsigned int lat[4][16 * LSTR32];   // packed bf16 pairs
  __shared__ float hstage[4][16 * HSTF];
  __shared__ float xstage[4][16 * XSTF];
  __shared__ float env_s[4][16];
  __shared__ int s_s[4][16];
  int lane = threadIdx.x & 31, wv = threadIdx.x >> 5;
  int e0 = (blockIdx.x * 4 + wv) * 16;
  int m = lane & 15, e = e0 + m;
  float* xg = ws + WS_X;
  float* Vg = ws + WS_V;
  const float* r0g = ws + WS_RHO0;
  const float* r1g = ws + WS_RHO1;
  int s = senders[e];
  if (lane < 16) { s_s[wv][m] = s; env_s[wv][m] = ws[WS_ENV + e]; }

  // Stage lat_in = [x | tp_scal | rho0_e] as packed bf16 pairs (each lane: one row-half)
  unsigned int* lrow = &lat[wv][m * LSTR32];
  if ((lane >> 4) == 0) {
    for (int c = 0; c < 64; c += 4) {
      float4 xv = *(const float4*)(xg + (size_t)e * 64 + c);
      lrow[(c >> 1)] = pk2(xv.x, xv.y);
      lrow[(c >> 1) + 1] = pk2(xv.z, xv.w);
    }
    for (int c = 0; c < 32; c += 4) {
      float a0 = 0, a1 = 0, a2 = 0, a3 = 0;
#pragma unroll
      for (int i = 0; i < 3; ++i) {
        float4 vv = *(const float4*)(Vg + ((size_t)e * 3 + i) * 64 + c);
        float4 rv = *(const float4*)(r1g + ((size_t)s * 3 + i) * 64 + c);
        a0 += vv.x * rv.x; a1 += vv.y * rv.y; a2 += vv.z * rv.z; a3 += vv.w * rv.w;
      }
      lrow[32 + (c >> 1)] = pk2(a0 * INV_AVG, a1 * INV_AVG);
      lrow[32 + (c >> 1) + 1] = pk2(a2 * INV_AVG, a3 * INV_AVG);
    }
  } else {
    for (int c = 32; c < 64; c += 4) {
      float a0 = 0, a1 = 0, a2 = 0, a3 = 0;
#pragma unroll
      for (int i = 0; i < 3; ++i) {
        float4 vv = *(const float4*)(Vg + ((size_t)e * 3 + i) * 64 + c);
        float4 rv = *(const float4*)(r1g + ((size_t)s * 3 + i) * 64 + c);
        a0 += vv.x * rv.x; a1 += vv.y * rv.y; a2 += vv.z * rv.z; a3 += vv.w * rv.w;
      }
      lrow[32 + (c >> 1)] = pk2(a0 * INV_AVG, a1 * INV_AVG);
      lrow[32 + (c >> 1) + 1] = pk2(a2 * INV_AVG, a3 * INV_AVG);
    }
    for (int c = 0; c < 64; c += 4) {
      float4 rv = *(const float4*)(r0g + (size_t)s * 64 + c);
      lrow[64 + (c >> 1)] = pk2(rv.x * INV_AVG, rv.y * INV_AVG);
      lrow[64 + (c >> 1) + 1] = pk2(rv.z * INV_AVG, rv.w * INV_AVG);
    }
  }
  // fused lat MLP: dx = silu(lat_in @ W1) @ W2
  v16bf a_lat[6];
#pragma unroll
  for (int kt = 0; kt < 6; ++kt) a_lat[kt] = load_A_ldsu(lat[wv], LSTR32, kt * 32, lane);
  v8f xacc[4] = {};
  for (int t = 0; t < 16; ++t) {
#pragma unroll
    for (int p = 0; p < 2; ++p) {
      v8f h = {};
#pragma unroll
      for (int kt = 0; kt < 6; ++kt)
        h = wmma_bf(a_lat[kt], load_B(wlat1, kt * 32 + 2 * t + p, lane), h);
#pragma unroll
      for (int r = 0; r < 8; ++r)
        hstage[wv][cm_row(r, lane) * HSTF + p * 16 + m] = silu(h[r]);
    }
    v16bf a_h = load_A_ldsf(hstage[wv], HSTF, 0, lane);
#pragma unroll
    for (int o = 0; o < 4; ++o)
      xacc[o] = wmma_bf(a_h, load_B(wlat2, t * 4 + o, lane), xacc[o]);
  }
  // residual latent update; restage new x (f32 LDS) for mix GEMM
  float evs[8];
#pragma unroll
  for (int r = 0; r < 8; ++r) evs[r] = env_s[wv][cm_row(r, lane)];
#pragma unroll
  for (int o = 0; o < 4; ++o)
#pragma unroll
    for (int r = 0; r < 8; ++r) {
      int mm = cm_row(r, lane);
      size_t xi = (size_t)(e0 + mm) * 64 + o * 16 + m;
      float xn = (xg[xi] + xacc[o][r] * evs[r]) * INV_SQRT2;
      xg[xi] = xn;
      xstage[wv][mm * XSTF + o * 16 + m] = xn;
    }
  // mix = x @ W_mix  (N=128 -> 8 accumulator tiles)
  v8f macc[8] = {};
#pragma unroll
  for (int kt = 0; kt < 2; ++kt) {
    v16bf a_x = load_A_ldsf(xstage[wv], XSTF, kt * 32, lane);
#pragma unroll
    for (int o = 0; o < 8; ++o)
      macc[o] = wmma_bf(a_x, load_B(wmix, kt * 8 + o, lane), macc[o]);
  }
  // V <- mix1*INV_AVG * (V*rho0) + mix2*INV_AVG * cross(V, rho1)   (INV_AVG folded by linearity)
  int sm[8];
#pragma unroll
  for (int r = 0; r < 8; ++r) sm[r] = s_s[wv][cm_row(r, lane)];
#pragma unroll
  for (int o = 0; o < 4; ++o)
#pragma unroll
    for (int r = 0; r < 8; ++r) {
      int mm = cm_row(r, lane);
      int c = o * 16 + m;
      size_t eb = (size_t)(e0 + mm) * 3;
      size_t sb = (size_t)sm[r] * 3;
      float v0 = Vg[(eb + 0) * 64 + c], v1 = Vg[(eb + 1) * 64 + c], v2 = Vg[(eb + 2) * 64 + c];
      float q0 = r1g[(sb + 0) * 64 + c];
      float q1 = r1g[(sb + 1) * 64 + c];
      float q2 = r1g[(sb + 2) * 64 + c];
      float p0 = r0g[(size_t)sm[r] * 64 + c];
      float mix1 = macc[o][r] * INV_AVG, mix2 = macc[o + 4][r] * INV_AVG;
      Vg[(eb + 0) * 64 + c] = mix1 * (v0 * p0) + mix2 * (v1 * q2 - v2 * q1);
      Vg[(eb + 1) * 64 + c] = mix1 * (v1 * p0) + mix2 * (v2 * q0 - v0 * q2);
      Vg[(eb + 2) * 64 + c] = mix1 * (v2 * p0) + mix2 * (v0 * q1 - v1 * q0);
    }
}

// -------- readout: e_edge = (x . W_out) * env, scatter to receivers --------
__global__ void k_out(const float* __restrict__ ws, const float* __restrict__ Wout,
                      const int* __restrict__ receivers, float* __restrict__ out) {
  int e = blockIdx.x * blockDim.x + threadIdx.x;
  if (e >= N_EDGES) return;
  const float* xr = ws + WS_X + (size_t)e * 64;
  float acc = 0.0f;
  for (int c = 0; c < 64; c += 4) {
    float4 a = *(const float4*)(xr + c);
    float4 w = *(const float4*)(Wout + c);
    acc += a.x * w.x + a.y * w.y + a.z * w.z + a.w * w.w;
  }
  acc *= ws[WS_ENV + e] * INV_AVG;
  atomicAdd(&out[receivers[e]], acc);
}

extern "C" void kernel_launch(void* const* d_in, const int* in_sizes, int n_in,
                              void* d_out, int out_size, void* d_ws, size_t ws_size,
                              hipStream_t stream) {
  (void)in_sizes; (void)n_in; (void)out_size; (void)ws_size;
  const float* pos   = (const float*)d_in[0];
  const float* Wemb1 = (const float*)d_in[1];
  const float* Wemb2 = (const float*)d_in[2];
  const float* Wv    = (const float*)d_in[3];
  const float* Wenv  = (const float*)d_in[4];
  const float* Wlat1 = (const float*)d_in[5];
  const float* Wlat2 = (const float*)d_in[6];
  const float* Wmix  = (const float*)d_in[7];
  const float* Wout  = (const float*)d_in[8];
  const int* species   = (const int*)d_in[9];
  const int* senders   = (const int*)d_in[10];
  const int* receivers = (const int*)d_in[11];
  float* ws = (float*)d_ws;
  unsigned int* wb = (unsigned int*)(ws + WS_WB);
  float* out = (float*)d_out;

  auto pack = [&](const float* W, int K, int N, unsigned off) {
    int tiles = ((K + 31) / 32) * (N / 16);
    int thr = tiles * 32;
    k_pack<<<(thr + 255) / 256, 256, 0, stream>>>(W, K, N, wb + off, tiles);
  };
  pack(Wemb1, 12, 512, WP_EMB1);
  pack(Wemb2, 512, 64, WP_EMB2);
  pack(Wv, 64, 64, WP_WV);
  for (int l = 0; l < 2; ++l) {
    pack(Wenv + (size_t)l * 64 * 64,    64, 64,  WP_WENV(l));
    pack(Wlat1 + (size_t)l * 192 * 512, 192, 512, WP_LAT1(l));
    pack(Wlat2 + (size_t)l * 512 * 64,  512, 64, WP_LAT2(l));
    pack(Wmix + (size_t)l * 64 * 128,   64, 128, WP_MIX(l));
  }
  k_zero<<<(N_NODES + 255) / 256, 256, 0, stream>>>(out, N_NODES);
  k_embed<<<N_EDGES / 64, 128, 0, stream>>>(pos, species, senders, receivers, wb, ws);
  for (int l = 0; l < 2; ++l) {
    k_zero<<<(2560000 + 255) / 256, 256, 0, stream>>>(ws + WS_RHO0, 2560000); // rho0+rho1
    k_env<<<N_EDGES / 64, 128, 0, stream>>>(ws + WS_X, ws + WS_U, senders,
                                            wb + WP_WENV(l), ws + WS_RHO0, ws + WS_RHO1);
    k_layer<<<N_EDGES / 64, 128, 0, stream>>>(ws, senders, wb + WP_LAT1(l),
                                              wb + WP_LAT2(l), wb + WP_MIX(l));
  }
  k_out<<<(N_EDGES + 255) / 256, 256, 0, stream>>>(ws, Wout, receivers, out);
}
